// F8Linear_25812753449730
// MI455X (gfx1250) — 8-bit matrix operand order repaired and checked by reading, compiled, not run on hardware
//
#include <hip/hip_runtime.h>


typedef __attribute__((ext_vector_type(16))) int   v16i;
typedef __attribute__((ext_vector_type(8)))  float v8f;

#define E4M3_MAX 448.0f
#define E5M2_MAX 57344.0f

#define K_DIM 4096
#define N_DIM 16384
#define M_DIM 8192

#define LDS_STRIDE 144
#define TILE_ROWS  128
#define TILE_BYTES (TILE_ROWS * LDS_STRIDE)
#define BUF_BYTES  (2 * TILE_BYTES)


__global__ void init_scales_kernel(unsigned int* s) {
    if (threadIdx.x == 0) { s[0] = 0u; s[1] = 0u; }
}

__global__ void amax_kernel(const float* __restrict__ in, size_t n4,
                            unsigned int* __restrict__ out_bits) {
    __shared__ unsigned int red[256];
    const float4* p = (const float4*)in;
    float m = 0.0f;
    for (size_t i = (size_t)blockIdx.x * blockDim.x + threadIdx.x; i < n4;
         i += (size_t)gridDim.x * blockDim.x) {
        float4 v = p[i];
        m = fmaxf(m, fmaxf(fmaxf(fabsf(v.x), fabsf(v.y)),
                           fmaxf(fabsf(v.z), fabsf(v.w))));
    }
    red[threadIdx.x] = __float_as_uint(m);
    __syncthreads();
    for (int s = 128; s > 0; s >>= 1) {
        if ((int)threadIdx.x < s) {
            unsigned int a = red[threadIdx.x], b = red[threadIdx.x + s];
            red[threadIdx.x] = (a > b) ? a : b;
        }
        __syncthreads();
    }
    if (threadIdx.x == 0) atomicMax(out_bits, red[0]);
}

__global__ void compute_scales_kernel(float* s) {
    if (threadIdx.x == 0) {
        float xa = __uint_as_float(((unsigned int*)s)[0]);
        float wa = __uint_as_float(((unsigned int*)s)[1]);
        float xs = fminf(E5M2_MAX / fmaxf(xa, 1e-12f), E5M2_MAX);
        float ws = fminf(E4M3_MAX / fmaxf(wa, 1e-12f), E4M3_MAX);
        s[2] = 1.0f / (xs * ws);
        s[3] = xs;
        s[4] = ws;
    }
}


__global__ void quant_bf8_kernel(const float* __restrict__ in,
                                 unsigned char* __restrict__ out,
                                 const float* __restrict__ scales, size_t n4) {
    const float s = scales[3];
    const float4* p = (const float4*)in;
    int* o = (int*)out;
    for (size_t i = (size_t)blockIdx.x * blockDim.x + threadIdx.x; i < n4;
         i += (size_t)gridDim.x * blockDim.x) {
        float4 v = p[i];
        float a = fminf(fmaxf(v.x * s, -E5M2_MAX), E5M2_MAX);
        float b = fminf(fmaxf(v.y * s, -E5M2_MAX), E5M2_MAX);
        float c = fminf(fmaxf(v.z * s, -E5M2_MAX), E5M2_MAX);
        float d = fminf(fmaxf(v.w * s, -E5M2_MAX), E5M2_MAX);
        int pk = 0;
        pk = __builtin_amdgcn_cvt_pk_bf8_f32(a, b, pk, false);
        pk = __builtin_amdgcn_cvt_pk_bf8_f32(c, d, pk, true);
        o[i] = pk;
    }
}

__global__ void quant_fp8_kernel(const float* __restrict__ in,
                                 unsigned char* __restrict__ out,
                                 const float* __restrict__ scales, size_t n4) {
    const float s = scales[4];
    const float4* p = (const float4*)in;
    int* o = (int*)out;
    for (size_t i = (size_t)blockIdx.x * blockDim.x + threadIdx.x; i < n4;
         i += (size_t)gridDim.x * blockDim.x) {
        float4 v = p[i];
        float a = fminf(fmaxf(v.x * s, -E4M3_MAX), E4M3_MAX);
        float b = fminf(fmaxf(v.y * s, -E4M3_MAX), E4M3_MAX);
        float c = fminf(fmaxf(v.z * s, -E4M3_MAX), E4M3_MAX);
        float d = fminf(fmaxf(v.w * s, -E4M3_MAX), E4M3_MAX);
        int pk = 0;
        pk = __builtin_amdgcn_cvt_pk_fp8_f32(a, b, pk, false);
        pk = __builtin_amdgcn_cvt_pk_fp8_f32(c, d, pk, true);
        o[i] = pk;
    }
}


__device__ __forceinline__ void async_ld16(unsigned lds_off,
                                           const unsigned char* g) {
    unsigned long long ga = (unsigned long long)(size_t)g;
    asm volatile("global_load_async_to_lds_b128 %0, %1, off"
                 :: "v"(lds_off), "v"(ga)
                 : "memory");
}

#define WAIT_ASYNC_8() asm volatile("s_wait_asynccnt 0x8" ::: "memory")
#define WAIT_ASYNC_0() asm volatile("s_wait_asynccnt 0x0" ::: "memory")


union AFrag { v16i v; unsigned long long u[8]; };
union BFrag { v16i v; int4 q[4]; };

__global__ __launch_bounds__(256) void f8_gemm_kernel(
    const unsigned char* __restrict__ xq,
    const unsigned char* __restrict__ wq,
    const float* __restrict__ bias,
    const float* __restrict__ scales,
    float* __restrict__ out) {
    __shared__ __align__(16) unsigned char smem[2][BUF_BYTES];

    const int tid   = threadIdx.x;
    const int lane  = tid & 31;
    const int wave  = tid >> 5;
    const int waveM = wave >> 1;
    const int waveN = wave & 1;
    const int l16   = lane & 15;
    const int hi    = lane >> 4;

    const int wgM = blockIdx.x * 128;
    const int wgN = blockIdx.y * 128;

    const int seg  = tid & 7;
    const int rowq = tid >> 3;

    const unsigned ldsBase = (unsigned)(size_t)&smem[0][0];

    const unsigned aOffL = (unsigned)((waveM * 32 + l16) * LDS_STRIDE + hi * 16);
    const unsigned bOffL = (unsigned)(TILE_BYTES +
                                      (waveN * 64 + l16) * LDS_STRIDE + hi * 16);

    v8f acc[2][4] = {};

    {
        const unsigned lbuf = ldsBase;
#pragma unroll
        for (int i = 0; i < 4; ++i) {
            int row = i * 32 + rowq;
            async_ld16(lbuf + row * LDS_STRIDE + seg * 16,
                       xq + (size_t)(wgM + row) * K_DIM + seg * 16);
        }
#pragma unroll
        for (int i = 0; i < 4; ++i) {
            int row = i * 32 + rowq;
            async_ld16(lbuf + TILE_BYTES + row * LDS_STRIDE + seg * 16,
                       wq + (size_t)(wgN + row) * K_DIM + seg * 16);
        }
    }

    for (int k0 = 0; k0 < K_DIM; k0 += 128) {
        const int buf = (k0 >> 7) & 1;
        const bool has_next = (k0 + 128) < K_DIM;

        if (has_next) {
            const unsigned lbuf = ldsBase + (buf ^ 1) * BUF_BYTES;
            const int kn = k0 + 128;
#pragma unroll
            for (int i = 0; i < 4; ++i) {
                int row = i * 32 + rowq;
                async_ld16(lbuf + row * LDS_STRIDE + seg * 16,
                           xq + (size_t)(wgM + row) * K_DIM + kn + seg * 16);
            }
#pragma unroll
            for (int i = 0; i < 4; ++i) {
                int row = i * 32 + rowq;
                async_ld16(lbuf + TILE_BYTES + row * LDS_STRIDE + seg * 16,
                           wq + (size_t)(wgN + row) * K_DIM + kn + seg * 16);
            }
            WAIT_ASYNC_8();
        } else {
            WAIT_ASYNC_0();
        }
        __syncthreads();

        const unsigned char* base = &smem[buf][0];

        AFrag a0, a1;
        {
            const unsigned long long* p0 =
                (const unsigned long long*)(base + aOffL);
            const unsigned long long* p1 =
                (const unsigned long long*)(base + aOffL + 16 * LDS_STRIDE);
#pragma unroll
            for (int j = 0; j < 8; ++j) { a0.u[j] = p0[4 * (j >> 1) + (j & 1)]; a1.u[j] = p1[4 * (j >> 1) + (j & 1)]; }
        }

#pragma unroll
        for (int n = 0; n < 4; ++n) {
            BFrag b;
            const int4* p = (const int4*)(base + bOffL + n * 16 * LDS_STRIDE);
#pragma unroll
            for (int j = 0; j < 4; ++j) b.q[j] = p[2 * j];
            acc[0][n] = __builtin_amdgcn_wmma_f32_16x16x128_bf8_fp8(
                a0.v, b.v, (short)0, acc[0][n], false, false);
            acc[1][n] = __builtin_amdgcn_wmma_f32_16x16x128_bf8_fp8(
                a1.v, b.v, (short)0, acc[1][n], false, false);
        }
        __syncthreads();
    }

    const float sc = scales[2];
#pragma unroll
    for (int n = 0; n < 4; ++n) {
        const int col = wgN + waveN * 64 + n * 16 + l16;
        const float bv = bias[col];
#pragma unroll
        for (int m = 0; m < 2; ++m) {
            const int rowTop = wgM + waveM * 32 + m * 16 + hi * 8;
            const v8f acv = acc[m][n];
#pragma unroll
            for (int r = 0; r < 8; ++r) {
                out[(size_t)(rowTop + r) * N_DIM + col] = acv[r] * sc + bv;
            }
        }
    }
}


extern "C" void kernel_launch(void* const* d_in, const int* in_sizes, int n_in,
                              void* d_out, int out_size, void* d_ws, size_t ws_size,
                              hipStream_t stream) {
    const float* x    = (const float*)d_in[0];
    const float* w    = (const float*)d_in[1];
    const float* bias = (const float*)d_in[2];
    float* out        = (float*)d_out;

    unsigned char* ws8 = (unsigned char*)d_ws;
    float* scales      = (float*)ws8;
    unsigned char* xq  = ws8 + 64;
    unsigned char* wq  = xq + (size_t)M_DIM * K_DIM;

    init_scales_kernel<<<1, 64, 0, stream>>>((unsigned int*)scales);
    amax_kernel<<<1024, 256, 0, stream>>>(x, (size_t)M_DIM * K_DIM / 4,
                                          (unsigned int*)scales + 0);
    amax_kernel<<<1024, 256, 0, stream>>>(w, (size_t)N_DIM * K_DIM / 4,
                                          (unsigned int*)scales + 1);
    compute_scales_kernel<<<1, 64, 0, stream>>>(scales);
    quant_bf8_kernel<<<2048, 256, 0, stream>>>(x, xq, scales,
                                               (size_t)M_DIM * K_DIM / 4);
    quant_fp8_kernel<<<2048, 256, 0, stream>>>(w, wq, scales,
                                               (size_t)N_DIM * K_DIM / 4);

    dim3 grid(M_DIM / 128, N_DIM / 128);
    f8_gemm_kernel<<<grid, 256, 0, stream>>>(xq, wq, bias, scales, out);
}
